// MutiHeadAttention_18279380811977
// MI455X (gfx1250) — matrix operand order repaired and checked by reading, not compiled, not run on hardware
//
#include <hip/hip_runtime.h>
#include <hip/hip_bf16.h>

// ---------- types ----------
typedef __bf16 bf16_t;
typedef bf16_t v16bf __attribute__((ext_vector_type(16)));
typedef bf16_t v8bf  __attribute__((ext_vector_type(8)));
typedef bf16_t v2bf  __attribute__((ext_vector_type(2)));
typedef float  v8f   __attribute__((ext_vector_type(8)));

union Frag16 {
    v16bf v;
    struct { v8bf lo; v8bf hi; } h;
};

// Problem constants (from reference): B=8, C=256 (head dim), L=1024, NH=8, D=2048
#define PK_B   8
#define PK_C   256
#define PK_L   1024
#define PK_NH  8
#define PK_D   2048

// ---------- optional CDNA5 async global->LDS path (guarded; falls back cleanly) ----------
#if defined(__AMDGCN__) && \
    __has_builtin(__builtin_amdgcn_global_load_async_to_lds_b128) && \
    __has_builtin(__builtin_amdgcn_s_wait_asynccnt)
#define USE_ASYNC_LDS 1
#else
#define USE_ASYNC_LDS 0
#endif

#if USE_ASYNC_LDS
// Builtin signature (from compiler diagnostic): params are pointers to
// GCC-vector 'int __vector(4)' in AS1 (global) / AS3 (LDS).
typedef int v4i_gcc __attribute__((vector_size(16)));
typedef __attribute__((address_space(1))) v4i_gcc* gv4i_t;
typedef __attribute__((address_space(3))) v4i_gcc* lv4i_t;
__device__ __forceinline__ gv4i_t to_global(const void* p) {
    return (gv4i_t)(unsigned long long)p;                 // inttoptr, AS1
}
__device__ __forceinline__ lv4i_t to_lds(const void* p) {
    // generic LDS addr low 32 bits == LDS offset (ISA 10.2 aperture rule)
    return (lv4i_t)(unsigned int)(unsigned long long)p;   // inttoptr, AS3
}
#endif

// =====================================================================
// Kernel 1: QKV projection.  y(L x 2048) = t(L x 256) @ W(256 x 2048) + b
// t[l][c] = x[b][c][l].  Output stored bf16 in (B, NH, L, 256) layout.
// Q is pre-scaled by 1/sqrt(C) so attention needs no per-score scaling.
// Block = 256 threads = 8 waves. WG tile: 64 tokens x 256 outputs.
// =====================================================================
#define TS_PAD 48   // row stride (elements) for LDS tiles; 96B = 32B-aligned rows

__global__ __launch_bounds__(256) void proj_qkv_kernel(
    const float* __restrict__ x,
    const float* __restrict__ Wq, const float* __restrict__ bq,
    const float* __restrict__ Wk, const float* __restrict__ bk,
    const float* __restrict__ Wv, const float* __restrict__ bv,
    bf16_t* __restrict__ qws, bf16_t* __restrict__ kws, bf16_t* __restrict__ vws)
{
    __shared__ __align__(32) bf16_t ts[64 * TS_PAD];    // token-major x tile (64 tok x 32 ch)
    __shared__ __align__(32) bf16_t wt[256 * TS_PAD];   // output-major W tile (256 out x 32 ch)

    const int t0 = blockIdx.x * 64;      // token tile base
    const int o0 = blockIdx.y * 256;     // output tile base
    const int zb = blockIdx.z;           // proj*8 + batch
    const int proj = zb >> 3;
    const int b    = zb & 7;

    const float* Wm = (proj == 0) ? Wq : (proj == 1) ? Wk : Wv;
    const float* bm = (proj == 0) ? bq : (proj == 1) ? bk : bv;
    bf16_t*     dst = (proj == 0) ? qws : (proj == 1) ? kws : vws;
    const float oscale = (proj == 0) ? 0.0625f : 1.0f;   // fold 1/sqrt(256) into Q

    const int tid  = threadIdx.x;
    const int wave = tid >> 5;
    const int lane = tid & 31;
    const int lm   = lane & 15;
    const int half = lane >> 4;

    const int mBase = (wave >> 2) * 32;  // token offset of this wave within the 64-row tile
    const int nBase = (wave & 3) * 64;   // output offset of this wave within the 256-col tile

    v8f acc[2][4];
    v8f vzero = {};
    #pragma unroll
    for (int i = 0; i < 2; ++i)
        #pragma unroll
        for (int j = 0; j < 4; ++j) acc[i][j] = vzero;

    const float* xb = x + (size_t)b * PK_C * PK_L;

    for (int k = 0; k < PK_C; k += 32) {
        __syncthreads();
        // stage x tile: 64 tokens x 32 channels, bf16, token-major (coalesced read along tokens)
        #pragma unroll
        for (int e = 0; e < 8; ++e) {
            int idx = e * 256 + tid;          // 0..2047
            int tt  = idx & 63;
            int cc  = idx >> 6;
            float v = xb[(size_t)(k + cc) * PK_L + t0 + tt];
            ts[tt * TS_PAD + cc] = (bf16_t)v;
        }
        // stage W tile: 256 outputs x 32 channels, bf16, output-major (coalesced read along outputs)
        #pragma unroll
        for (int e = 0; e < 32; ++e) {
            int idx = e * 256 + tid;          // 0..8191
            int oo  = idx & 255;
            int cc  = idx >> 8;
            float v = Wm[(size_t)(k + cc) * PK_D + o0 + oo];
            wt[oo * TS_PAD + cc] = (bf16_t)v;
        }
        __syncthreads();

        // A fragments (16x32 bf16): lane row = lm, 16 contiguous K at offset half*16
        Frag16 afr[2];
        #pragma unroll
        for (int mi = 0; mi < 2; ++mi) {
            const bf16_t* base = &ts[(mBase + mi * 16 + lm) * TS_PAD + half * 16];
            afr[mi].h.lo = *(const v8bf*)(base);
            afr[mi].h.hi = *(const v8bf*)(base + 8);
        }
        // B fragments (32x16 bf16): lane col = lm, 16 contiguous K at offset half*16
        #pragma unroll
        for (int ni = 0; ni < 4; ++ni) {
            const bf16_t* bb = &wt[(nBase + ni * 16 + lm) * TS_PAD + half * 16];
            v16bf bfrag = *(const v16bf*)bb;
            #pragma unroll
            for (int mi = 0; mi < 2; ++mi) {
                acc[mi][ni] = __builtin_amdgcn_wmma_f32_16x16x32_bf16(
                    false, afr[mi].v, false, bfrag, (short)0, acc[mi][ni], false, false);
            }
        }
    }

    // epilogue: add bias, apply oscale, convert bf16, scatter into (B, NH, L, 256) layout
    #pragma unroll
    for (int ni = 0; ni < 4; ++ni) {
        int o = o0 + nBase + ni * 16 + lm;    // global output column 0..2047
        float bias = bm[o];
        int hh = o >> 8;                      // head
        int dh = o & 255;                     // channel within head
        bf16_t* drow = dst + ((size_t)(b * PK_NH + hh) * PK_L) * PK_C + dh;
        #pragma unroll
        for (int mi = 0; mi < 2; ++mi) {
            int tokBase = t0 + mBase + mi * 16 + half * 8;
            #pragma unroll
            for (int r = 0; r < 8; ++r) {
                float v = (acc[mi][ni][r] + bias) * oscale;
                drow[(size_t)(tokBase + r) * PK_C] = (bf16_t)v;
            }
        }
    }
}

// =====================================================================
// Kernel 2: flash attention per (b,h).  Block = 256 threads = 8 waves.
// WG handles a 128-row Q block; wave w owns rows [w*16, w*16+16).
// Iterates keys in blocks of 32: S = Q K^T (2 tiles of 16x16, K=256),
// online softmax, O += P V with P round-tripped through per-wave LDS.
// K block staged via GLOBAL_LOAD_ASYNC_TO_LDS when available.
// =====================================================================
__global__ __launch_bounds__(256) void attn_kernel(
    const bf16_t* __restrict__ qws, const bf16_t* __restrict__ kws,
    const bf16_t* __restrict__ vws, float* __restrict__ out)
{
    __shared__ __align__(32) bf16_t Ks[32 * 256];     // key-major K block (32 keys x 256 ch)
    __shared__ __align__(32) bf16_t Vt[256 * 32];     // channel-major V block (256 ch x 32 keys)
    __shared__ __align__(32) bf16_t Pb[8][16 * 32];   // per-wave P tile (16 rows x 32 keys)

    const int bh = blockIdx.y;            // 0..63  (= b*8 + h)
    const int q0 = blockIdx.x * 128;      // Q block base
    const int b  = bh >> 3;
    const int h  = bh & 7;

    const int tid  = threadIdx.x;
    const int wave = tid >> 5;
    const int lane = tid & 31;
    const int lm   = lane & 15;
    const int half = lane >> 4;

    const bf16_t* qh = qws + (size_t)bh * PK_L * PK_C;
    const bf16_t* kh = kws + (size_t)bh * PK_L * PK_C;
    const bf16_t* vh = vws + (size_t)bh * PK_L * PK_C;

    // Q strip fragments kept in registers: row = q0 + wave*16 + lm, 8 K-steps of 32
    Frag16 qf[8];
    {
        const bf16_t* qrow = qh + (size_t)(q0 + wave * 16 + lm) * PK_C;
        #pragma unroll
        for (int c = 0; c < 8; ++c) {
            qf[c].h.lo = *(const v8bf*)(qrow + c * 32 + half * 16);
            qf[c].h.hi = *(const v8bf*)(qrow + c * 32 + half * 16 + 8);
        }
    }

    v8f accO[16];
    v8f vzero = {};
    #pragma unroll
    for (int i = 0; i < 16; ++i) accO[i] = vzero;

    float mrow[8], lrow[8];
    #pragma unroll
    for (int r = 0; r < 8; ++r) { mrow[r] = -1e30f; lrow[r] = 0.0f; }

    for (int jb = 0; jb < PK_L; jb += 32) {
        __syncthreads();
        // stage K block (natural layout), 16KB
#if USE_ASYNC_LDS
        {
            const bf16_t* src = kh + (size_t)jb * PK_C;
            #pragma unroll
            for (int e = 0; e < 4; ++e) {
                size_t chunk = (size_t)(e * 256 + tid) * 8;   // 8 bf16 = 16 bytes
                __builtin_amdgcn_global_load_async_to_lds_b128(
                    to_global(src + chunk), to_lds(Ks + chunk), 0, 0);
            }
        }
#else
        {
            const uint4* s4 = (const uint4*)(kh + (size_t)jb * PK_C);
            uint4*       d4 = (uint4*)Ks;
            #pragma unroll
            for (int e = 0; e < 4; ++e) d4[e * 256 + tid] = s4[e * 256 + tid];
        }
#endif
        // stage V block transposed: Vt[ch][key]; thread owns channel = tid
        {
            const bf16_t* src = vh + (size_t)jb * PK_C;
            #pragma unroll
            for (int kp = 0; kp < 16; ++kp) {
                bf16_t v0 = src[(size_t)(2 * kp) * PK_C + tid];
                bf16_t v1 = src[(size_t)(2 * kp + 1) * PK_C + tid];
                v2bf pk = {v0, v1};
                *(v2bf*)&Vt[tid * 32 + 2 * kp] = pk;
            }
        }
#if USE_ASYNC_LDS
        __builtin_amdgcn_s_wait_asynccnt(0);
#endif
        __syncthreads();

        // S tiles: s[t] = Q(16x256) . K^T for keys jb+t*16 .. jb+t*16+15
        v8f s[2];
        s[0] = vzero; s[1] = vzero;
        #pragma unroll
        for (int t = 0; t < 2; ++t) {
            #pragma unroll
            for (int c = 0; c < 8; ++c) {
                const bf16_t* kb = &Ks[(t * 16 + lm) * 256 + c * 32 + half * 16];
                v16bf bfrag = *(const v16bf*)kb;
                s[t] = __builtin_amdgcn_wmma_f32_16x16x32_bf16(
                    false, qf[c].v, false, bfrag, (short)0, s[t], false, false);
            }
        }

        // online softmax (row = half*8 + r lives across the 16 lanes of this half)
        #pragma unroll
        for (int r = 0; r < 8; ++r) {
            float s0 = s[0][r];               // Q already carries 1/sqrt(C)
            float s1 = s[1][r];
            float mx = fmaxf(s0, s1);
            #pragma unroll
            for (int m = 8; m >= 1; m >>= 1) mx = fmaxf(mx, __shfl_xor(mx, m, 32));
            float mnew = fmaxf(mrow[r], mx);
            float corr = __expf(mrow[r] - mnew);
            float p0 = __expf(s0 - mnew);
            float p1 = __expf(s1 - mnew);
            float ps = p0 + p1;
            #pragma unroll
            for (int m = 8; m >= 1; m >>= 1) ps += __shfl_xor(ps, m, 32);
            lrow[r] = lrow[r] * corr + ps;
            mrow[r] = mnew;
            // rescale existing O row
            #pragma unroll
            for (int ct = 0; ct < 16; ++ct) accO[ct][r] *= corr;
            // write P row into per-wave LDS tile (row = half*8+r, keys tile*16+lm)
            Pb[wave][(half * 8 + r) * 32 + lm]      = (bf16_t)p0;
            Pb[wave][(half * 8 + r) * 32 + 16 + lm] = (bf16_t)p1;
        }

        // reload P as an A fragment (16 rows x 32 keys)
        Frag16 pf;
        {
            const bf16_t* pr = &Pb[wave][lm * 32 + half * 16];
            pf.h.lo = *(const v8bf*)(pr);
            pf.h.hi = *(const v8bf*)(pr + 8);
        }

        // O(16x256) += P(16x32) . V(32x256): one WMMA per 16-channel tile
        #pragma unroll
        for (int ct = 0; ct < 16; ++ct) {
            const bf16_t* vb = &Vt[(ct * 16 + lm) * 32 + half * 16];
            v16bf bfrag = *(const v16bf*)vb;
            accO[ct] = __builtin_amdgcn_wmma_f32_16x16x32_bf16(
                false, pf.v, false, bfrag, (short)0, accO[ct], false, false);
        }
    }

    // epilogue: normalize and store fp32 to (B, L, NH*256)
    float inv[8];
    #pragma unroll
    for (int r = 0; r < 8; ++r) inv[r] = 1.0f / lrow[r];

    float* obase = out + ((size_t)b * PK_L + q0 + wave * 16) * PK_D + h * PK_C;
    #pragma unroll
    for (int ct = 0; ct < 16; ++ct) {
        int ch = ct * 16 + lm;
        #pragma unroll
        for (int r = 0; r < 8; ++r) {
            int tok = half * 8 + r;
            obase[(size_t)tok * PK_D + ch] = accO[ct][r] * inv[r];
        }
    }
}

// =====================================================================
// launch
// =====================================================================
extern "C" void kernel_launch(void* const* d_in, const int* in_sizes, int n_in,
                              void* d_out, int out_size, void* d_ws, size_t ws_size,
                              hipStream_t stream) {
    const float* x  = (const float*)d_in[0];
    const float* Wq = (const float*)d_in[1];
    const float* bq = (const float*)d_in[2];
    const float* Wk = (const float*)d_in[3];
    const float* bk = (const float*)d_in[4];
    const float* Wv = (const float*)d_in[5];
    const float* bv = (const float*)d_in[6];

    const size_t perTensor = (size_t)PK_B * PK_NH * PK_L * PK_C; // 16,777,216 bf16 elems
    bf16_t* qws = (bf16_t*)d_ws;
    bf16_t* kws = qws + perTensor;
    bf16_t* vws = kws + perTensor;

    // projections: grid (token tiles=16, output tiles=8, proj*batch=24)
    proj_qkv_kernel<<<dim3(PK_L / 64, PK_D / 256, 3 * PK_B), 256, 0, stream>>>(
        x, Wq, bq, Wk, bk, Wv, bv, qws, kws, vws);

    // attention: grid (q blocks=8, batch-heads=64)
    attn_kernel<<<dim3(PK_L / 128, PK_B * PK_NH), 256, 0, stream>>>(
        qws, kws, vws, (float*)d_out);
}
